// Baseline_35622458753505
// MI455X (gfx1250) — matrix operand order repaired and checked by reading, not compiled, not run on hardware
//
#include <hip/hip_runtime.h>
#include <math.h>

// ---------------------------------------------------------------------------
// Problem sizes (match reference)
// ---------------------------------------------------------------------------
constexpr int B_  = 32;
constexpr int S_  = 512;
constexpr int E_  = 768;
constexpr int H_  = 768;
constexpr int T_  = 9;
constexpr int G4H = 4 * H_;        // 3072
constexpr int BS  = B_ * S_;       // 16384

// ---------------------------------------------------------------------------
// WMMA vector types (CDNA5 gfx1250, wave32)
// ---------------------------------------------------------------------------
typedef __attribute__((ext_vector_type(16))) __bf16       v16bf;
typedef __attribute__((ext_vector_type(8)))  float        v8f;
typedef __attribute__((ext_vector_type(4)))  unsigned int u32x4;

union ABFrag { v16bf v; u32x4 u[2]; };

// ---------------------------------------------------------------------------
// f32 -> bf16 conversion (weights)
// ---------------------------------------------------------------------------
__global__ void cvt_bf16_kernel(const float* __restrict__ src,
                                __bf16* __restrict__ dst, int n) {
  int i = blockIdx.x * blockDim.x + threadIdx.x;
  if (i < n) dst[i] = (__bf16)src[i];
}

// Embedding gather + bf16 convert: X[b*S+s, :] = emb[ids[b*S+s], :]
__global__ void embed_kernel(const int* __restrict__ ids,
                             const float* __restrict__ emb,
                             __bf16* __restrict__ X) {
  int i = blockIdx.x * blockDim.x + threadIdx.x;
  if (i >= BS * E_) return;
  int row = i / E_, col = i - row * E_;
  X[i] = (__bf16)emb[(long)ids[row] * E_ + col];
}

__global__ void zero_state_kernel(__bf16* __restrict__ hbf,
                                  float* __restrict__ c) {
  int i = blockIdx.x * blockDim.x + threadIdx.x;
  if (i < B_ * H_) { hbf[i] = (__bf16)0.0f; c[i] = 0.0f; }
}

// ---------------------------------------------------------------------------
// Big input GEMM with 4-way A-fragment reuse:
//   Y[s*B+b, n] = sum_k A[b*S+s, k] * W[n, k] + bias0[n] + bias1[n]
// Each wave owns one 16-row M tile and 4 consecutive 16-col N tiles:
// one A-fragment load feeds 4 v_wmma_f32_16x16x32_bf16 per K-step.
// Output permuted to time-major [S, B, 4H] for the recurrence.
// ---------------------------------------------------------------------------
constexpr int NT = 4;   // N tiles per wave

__global__ __launch_bounds__(256)
void gemm_xg_kernel(const __bf16* __restrict__ A,   // [M, K] bf16, row = b*S+s
                    const __bf16* __restrict__ W,   // [N, K] bf16
                    const float* __restrict__ b0,
                    const float* __restrict__ b1,
                    float* __restrict__ Y,          // [S, B, N] time-major
                    int M, int N, int K) {
  const int wave = threadIdx.x >> 5;
  const int lane = threadIdx.x & 31;
  const int groups_n = (N >> 4) / NT;
  const long grp   = (long)blockIdx.x * 8 + wave;
  const long total = (long)(M >> 4) * groups_n;
  if (grp >= total) return;
  const int gn = (int)(grp % groups_n);
  const int tm = (int)(grp / groups_n);
  const int m0 = tm << 4;
  const int n0 = (gn * NT) << 4;

  // A fragment: lane (l&15) = row M; hi lanes take K+8 / K+24 halves
  const int mrow = m0 + (lane & 15);
  const int kba  = (lane >> 4) << 3;          // 0 or 8
  // B fragment: lane (l&15) = col N; hi lanes take K+8 / K+24 halves
  const int ncol = lane & 15;
  const int kbb  = (lane >> 4) << 3;          // 0 or 8

  const __bf16* pa = A + (long)mrow * K + kba;
  const __bf16* pb = W + (long)(n0 + ncol) * K + kbb;
  const long nstride = (long)16 * K;          // between N tiles

  v8f acc[NT];
#pragma unroll
  for (int nt = 0; nt < NT; ++nt) acc[nt] = (v8f){};

  for (int k = 0; k < K; k += 32) {
    ABFrag a;
    a.u[0] = *(const u32x4*)(const void*)(pa);        // K = kba..kba+7
    a.u[1] = *(const u32x4*)(const void*)(pa + 16);   // K = kba+16..kba+23
    __builtin_prefetch(pa + 32, 0, 1);
#pragma unroll
    for (int nt = 0; nt < NT; ++nt) {
      ABFrag b;
      const __bf16* p = pb + nt * nstride;
      b.u[0] = *(const u32x4*)(const void*)(p);       // K = kbb..kbb+7
      b.u[1] = *(const u32x4*)(const void*)(p + 16);  // K = kbb+16..kbb+23
      __builtin_prefetch(p + 32, 0, 1);
      acc[nt] = __builtin_amdgcn_wmma_f32_16x16x32_bf16(
          false, a.v, false, b.v, (short)0, acc[nt], false, false);
    }
    pa += 32; pb += 32;
  }

  const int rbase = (lane >> 4) << 3;  // C/D: VGPR r -> M = r (+8 for hi lanes)
#pragma unroll
  for (int nt = 0; nt < NT; ++nt) {
    const int col = n0 + nt * 16 + ncol;
    const float bias = b0[col] + b1[col];
    for (int r = 0; r < 8; ++r) {
      const int row = m0 + rbase + r;          // row = b*S + s
      const int bi = row / S_, s = row - bi * S_;
      Y[((long)s * B_ + bi) * N + col] = acc[nt][r] + bias;
    }
  }
}

// ---------------------------------------------------------------------------
// Recurrent step GEMM:  G[b, n] = Xg[t, b, n] + sum_k h[b,k]*Whh[n,k]
// h (32x768 bf16, 48KB) is staged into LDS once per block via the gfx1250
// async-to-LDS path (ASYNCcnt), then each wave computes one 16-col N tile
// for BOTH 16-row M tiles, reusing the weight fragment for 2 WMMAs.
// Grid: 192 N tiles / 8 waves = 24 blocks.
// ---------------------------------------------------------------------------
__global__ __launch_bounds__(256)
void gemm_step_kernel(const __bf16* __restrict__ hbf,  // [32, K]
                      const __bf16* __restrict__ W,    // [N, K]
                      const float* __restrict__ Xg,    // [S, B, N]
                      float* __restrict__ G,           // [32, N]
                      int t, int N, int K) {
  __shared__ __bf16 hs[B_ * H_];                       // 48 KB
  const int tid = threadIdx.x;

  // ---- async stage h -> LDS (global_load_async_to_lds_b128, ASYNCcnt) ----
  {
    const unsigned ldsbase = (unsigned)(size_t)(void*)hs;
    const int nchunks = (B_ * H_) / 8;                 // 16B chunks = 3072
    for (int i = tid; i < nchunks; i += 256) {
      const __bf16* src = hbf + (long)i * 8;
      const unsigned dst = ldsbase + (unsigned)i * 16u;
      asm volatile("global_load_async_to_lds_b128 %0, %1, off"
                   :: "v"(dst), "v"(src) : "memory");
    }
    asm volatile("s_wait_asynccnt 0x0" ::: "memory");
  }
  __syncthreads();

  const int wave = tid >> 5;
  const int lane = tid & 31;
  const int tiles_n = N >> 4;
  const int tn = blockIdx.x * 8 + wave;
  if (tn >= tiles_n) return;
  const int n0 = tn << 4;

  const int ncol = lane & 15;
  const int kbb  = (lane >> 4) << 3;
  const int kba  = (lane >> 4) << 3;
  const int arow = lane & 15;

  const __bf16* pb  = W + (long)(n0 + ncol) * K + kbb;
  const __bf16* la0 = hs + (long)arow * K + kba;          // M tile 0
  const __bf16* la1 = hs + (long)(16 + arow) * K + kba;   // M tile 1

  v8f acc0 = {}, acc1 = {};
  for (int k = 0; k < K; k += 32) {
    ABFrag b, a0, a1;
    b.u[0]  = *(const u32x4*)(const void*)(pb);
    b.u[1]  = *(const u32x4*)(const void*)(pb + 16);
    a0.u[0] = *(const u32x4*)(const void*)(la0);       // ds_load_b128
    a0.u[1] = *(const u32x4*)(const void*)(la0 + 16);
    a1.u[0] = *(const u32x4*)(const void*)(la1);
    a1.u[1] = *(const u32x4*)(const void*)(la1 + 16);
    __builtin_prefetch(pb + 32, 0, 1);
    acc0 = __builtin_amdgcn_wmma_f32_16x16x32_bf16(
        false, a0.v, false, b.v, (short)0, acc0, false, false);
    acc1 = __builtin_amdgcn_wmma_f32_16x16x32_bf16(
        false, a1.v, false, b.v, (short)0, acc1, false, false);
    pb += 32; la0 += 32; la1 += 32;
  }

  const int rbase = (lane >> 4) << 3;
  const int col   = n0 + ncol;
  const float* xg = Xg + ((long)t * B_) * N;
  for (int r = 0; r < 8; ++r) {
    const int row0 = rbase + r;            // batch 0..15
    const int row1 = 16 + rbase + r;       // batch 16..31
    G[(long)row0 * N + col] = acc0[r] + xg[(long)row0 * N + col];
    G[(long)row1 * N + col] = acc1[r] + xg[(long)row1 * N + col];
  }
}

// ---------------------------------------------------------------------------
// LSTM gate nonlinearity + state update (torch gate order i,f,g,o)
// ---------------------------------------------------------------------------
__global__ void lstm_gate_kernel(const float* __restrict__ G,  // [32, 3072]
                                 float* __restrict__ c,        // [32, 768]
                                 __bf16* __restrict__ hbf,     // [32, 768]
                                 __bf16* __restrict__ Hbf,     // [BS,768] or null
                                 float*  __restrict__ Hf,      // [BS,768] or null
                                 int t) {
  int idx = blockIdx.x * blockDim.x + threadIdx.x;
  if (idx >= B_ * H_) return;
  const int b = idx / H_, j = idx - b * H_;
  const float* g = G + (long)b * G4H;
  const float gi = g[j], gf = g[H_ + j], gg = g[2 * H_ + j], go = g[3 * H_ + j];
  const float ii = 1.0f / (1.0f + __expf(-gi));
  const float ff = 1.0f / (1.0f + __expf(-gf));
  const float oo = 1.0f / (1.0f + __expf(-go));
  const float cc = ff * c[idx] + ii * tanhf(gg);
  const float hh = oo * tanhf(cc);
  c[idx]   = cc;
  hbf[idx] = (__bf16)hh;                      // next-step WMMA A operand
  const long orow = (long)b * S_ + t;
  if (Hbf) Hbf[orow * H_ + j] = (__bf16)hh;   // layer-0 output -> layer-1 input
  if (Hf)  Hf[orow * H_ + j]  = hh;           // layer-1 output -> logits
}

// ---------------------------------------------------------------------------
// Logits (N=9, too small for WMMA) + log_softmax, one thread per token row
// ---------------------------------------------------------------------------
__global__ void logits_logsoftmax_kernel(const float* __restrict__ Hf,   // [BS,768]
                                         const float* __restrict__ Wout, // [9,768]
                                         const float* __restrict__ bout, // [9]
                                         float* __restrict__ logp) {     // [BS,9]
  int r = blockIdx.x * blockDim.x + threadIdx.x;
  if (r >= BS) return;
  const float* h = Hf + (long)r * H_;
  float lg[T_];
  float mx = -1e30f;
  for (int tt = 0; tt < T_; ++tt) {
    const float* w = Wout + tt * H_;
    float s = bout[tt];
    for (int k = 0; k < H_; ++k) s += h[k] * w[k];
    lg[tt] = s; mx = fmaxf(mx, s);
  }
  float se = 0.0f;
  for (int tt = 0; tt < T_; ++tt) se += __expf(lg[tt] - mx);
  const float lse = mx + __logf(se);
  for (int tt = 0; tt < T_; ++tt) logp[(long)r * T_ + tt] = lg[tt] - lse;
}

// ---------------------------------------------------------------------------
// CRF token-mean NLL: single block, 288 threads = (b,t'), score in LDS
// ---------------------------------------------------------------------------
__global__ void crf_nll_kernel(const float* __restrict__ logp,   // [B,S,T]
                               const int* __restrict__ labels,   // [B,S]
                               const int* __restrict__ amask,    // [B,S]
                               const float* __restrict__ start_t,
                               const float* __restrict__ end_t,
                               const float* __restrict__ trans,  // [T,T]
                               float* __restrict__ out_loss) {
  __shared__ float score[B_][T_ + 1];
  __shared__ float nxt[B_][T_ + 1];
  __shared__ float tr[T_][T_];
  __shared__ float num[B_], denom[B_], msum[B_];
  const int tid = threadIdx.x;
  if (tid < T_ * T_) tr[tid / T_][tid % T_] = trans[tid];
  const int b = tid / T_, tt = tid - b * T_;
  __syncthreads();
  score[b][tt] = start_t[tt] + logp[((long)b * S_) * T_ + tt];
  __syncthreads();
  for (int s = 1; s < S_; ++s) {
    float m = -1e30f;
    for (int p = 0; p < T_; ++p) m = fmaxf(m, score[b][p] + tr[p][tt]);
    float e = 0.0f;
    for (int p = 0; p < T_; ++p) e += __expf(score[b][p] + tr[p][tt] - m);
    const float v = m + __logf(e) + logp[((long)b * S_ + s) * T_ + tt];
    nxt[b][tt] = amask[b * S_ + s] ? v : score[b][tt];
    __syncthreads();
    score[b][tt] = nxt[b][tt];
    __syncthreads();
  }
  if (tid < B_) {
    float m = -1e30f;
    for (int p = 0; p < T_; ++p) m = fmaxf(m, score[tid][p] + end_t[p]);
    float e = 0.0f;
    for (int p = 0; p < T_; ++p) e += __expf(score[tid][p] + end_t[p] - m);
    denom[tid] = m + __logf(e);
    // numerator + mask token count
    const int tag0 = labels[tid * S_];
    float n = start_t[tag0] + logp[((long)tid * S_) * T_ + tag0];
    int prev = tag0, last = tag0;
    float ms = (float)amask[tid * S_];
    for (int s = 1; s < S_; ++s) {
      const int mk = amask[tid * S_ + s];
      const int tg = labels[tid * S_ + s];
      if (mk) { n += tr[prev][tg] + logp[((long)tid * S_ + s) * T_ + tg];
                last = tg; ms += 1.0f; }
      prev = tg;
    }
    n += end_t[last];
    num[tid] = n; msum[tid] = ms;
  }
  __syncthreads();
  if (tid == 0) {
    float sll = 0.0f, sm = 0.0f;
    for (int bb = 0; bb < B_; ++bb) { sll += num[bb] - denom[bb]; sm += msum[bb]; }
    out_loss[0] = -sll / sm;
  }
}

// ---------------------------------------------------------------------------
// CRF Viterbi decode: single block, 288 threads; backtrace via uint8 history
// ---------------------------------------------------------------------------
__global__ void crf_viterbi_kernel(const float* __restrict__ logp,  // [B,S,T]
                                   const int* __restrict__ amask,   // [B,S]
                                   const float* __restrict__ start_t,
                                   const float* __restrict__ end_t,
                                   const float* __restrict__ trans, // [T,T]
                                   unsigned char* __restrict__ hist,// [S-1,B,T]
                                   float* __restrict__ out_preds) { // [B,S]
  __shared__ float score[B_][T_ + 1];
  __shared__ float nxt[B_][T_ + 1];
  __shared__ float tr[T_][T_];
  __shared__ int best[B_];
  const int tid = threadIdx.x;
  if (tid < T_ * T_) tr[tid / T_][tid % T_] = trans[tid];
  const int b = tid / T_, tt = tid - b * T_;
  __syncthreads();
  score[b][tt] = start_t[tt] + logp[((long)b * S_) * T_ + tt];
  __syncthreads();
  for (int s = 1; s < S_; ++s) {
    float m = -1e30f; int am = 0;
    for (int p = 0; p < T_; ++p) {
      const float v = score[b][p] + tr[p][tt];
      if (v > m) { m = v; am = p; }
    }
    const int mk = amask[b * S_ + s];
    nxt[b][tt] = mk ? (m + logp[((long)b * S_ + s) * T_ + tt]) : score[b][tt];
    hist[(long)(s - 1) * (B_ * T_) + b * T_ + tt] = (unsigned char)(mk ? am : tt);
    __syncthreads();
    score[b][tt] = nxt[b][tt];
    __syncthreads();
  }
  if (tid < B_) {
    float m = -1e30f; int am = 0;
    for (int p = 0; p < T_; ++p) {
      const float v = score[tid][p] + end_t[p];
      if (v > m) { m = v; am = p; }
    }
    best[tid] = am;
  }
  __syncthreads();
  if (tid < B_) {
    int tag = best[tid];
    out_preds[(long)tid * S_ + (S_ - 1)] =
        (float)(tag * amask[tid * S_ + (S_ - 1)]);
    for (int s = S_ - 2; s >= 0; --s) {
      tag = hist[(long)s * (B_ * T_) + tid * T_ + tag];
      out_preds[(long)tid * S_ + s] = (float)(tag * amask[tid * S_ + s]);
    }
  }
}

// ---------------------------------------------------------------------------
// Host orchestration
// ---------------------------------------------------------------------------
extern "C" void kernel_launch(void* const* d_in, const int* in_sizes, int n_in,
                              void* d_out, int out_size, void* d_ws, size_t ws_size,
                              hipStream_t stream) {
  (void)in_sizes; (void)n_in; (void)out_size; (void)ws_size;
  const int*   ids    = (const int*)d_in[0];
  const int*   amask  = (const int*)d_in[1];
  const int*   labels = (const int*)d_in[2];
  const float* emb    = (const float*)d_in[3];
  const float* Wih0   = (const float*)d_in[4];
  const float* Whh0   = (const float*)d_in[5];
  const float* bih0   = (const float*)d_in[6];
  const float* bhh0   = (const float*)d_in[7];
  const float* Wih1   = (const float*)d_in[8];
  const float* Whh1   = (const float*)d_in[9];
  const float* bih1   = (const float*)d_in[10];
  const float* bhh1   = (const float*)d_in[11];
  const float* Wout   = (const float*)d_in[12];
  const float* bout   = (const float*)d_in[13];
  const float* startt = (const float*)d_in[14];
  const float* endt   = (const float*)d_in[15];
  const float* trans  = (const float*)d_in[16];

  char* ws = (char*)d_ws;
  size_t off = 0;
  auto alloc = [&](size_t bytes) -> void* {
    void* p = ws + off;
    off = (off + bytes + 255) & ~(size_t)255;
    return p;
  };

  __bf16* Xbf   = (__bf16*)alloc((size_t)BS * E_ * 2);
  __bf16* Wih0b = (__bf16*)alloc((size_t)G4H * E_ * 2);
  __bf16* Whh0b = (__bf16*)alloc((size_t)G4H * H_ * 2);
  __bf16* Wih1b = (__bf16*)alloc((size_t)G4H * H_ * 2);
  __bf16* Whh1b = (__bf16*)alloc((size_t)G4H * H_ * 2);
  float*  Xg    = (float*)alloc((size_t)BS * G4H * 4);     // time-major [S,B,4H]
  float*  G     = (float*)alloc((size_t)B_ * G4H * 4);
  __bf16* hbf   = (__bf16*)alloc((size_t)B_ * H_ * 2);
  float*  c     = (float*)alloc((size_t)B_ * H_ * 4);
  __bf16* H0b   = (__bf16*)alloc((size_t)BS * H_ * 2);
  float*  H1f   = (float*)alloc((size_t)BS * H_ * 4);
  float*  logp  = (float*)alloc((size_t)BS * T_ * 4);
  unsigned char* hist = (unsigned char*)alloc((size_t)S_ * B_ * T_);

  const int nW = G4H * E_;   // all weight mats are 3072x768
  cvt_bf16_kernel<<<(nW + 255) / 256, 256, 0, stream>>>(Wih0, Wih0b, nW);
  cvt_bf16_kernel<<<(nW + 255) / 256, 256, 0, stream>>>(Whh0, Whh0b, nW);
  cvt_bf16_kernel<<<(nW + 255) / 256, 256, 0, stream>>>(Wih1, Wih1b, nW);
  cvt_bf16_kernel<<<(nW + 255) / 256, 256, 0, stream>>>(Whh1, Whh1b, nW);
  embed_kernel<<<(BS * E_ + 255) / 256, 256, 0, stream>>>(ids, emb, Xbf);

  const int gemmBlocks = ((BS / 16) * ((G4H / 16) / NT)) / 8;  // 6144
  const int stepBlocks = (G4H / 16) / 8;                       // 24
  const int gateBlocks = (B_ * H_ + 255) / 256;                // 96

  // ---- Layer 0 ----
  gemm_xg_kernel<<<gemmBlocks, 256, 0, stream>>>(Xbf, Wih0b, bih0, bhh0,
                                                 Xg, BS, G4H, E_);
  zero_state_kernel<<<gateBlocks, 256, 0, stream>>>(hbf, c);
  for (int t = 0; t < S_; ++t) {
    gemm_step_kernel<<<stepBlocks, 256, 0, stream>>>(hbf, Whh0b, Xg, G, t, G4H, H_);
    lstm_gate_kernel<<<gateBlocks, 256, 0, stream>>>(G, c, hbf, H0b, nullptr, t);
  }

  // ---- Layer 1 ----
  gemm_xg_kernel<<<gemmBlocks, 256, 0, stream>>>(H0b, Wih1b, bih1, bhh1,
                                                 Xg, BS, G4H, H_);
  zero_state_kernel<<<gateBlocks, 256, 0, stream>>>(hbf, c);
  for (int t = 0; t < S_; ++t) {
    gemm_step_kernel<<<stepBlocks, 256, 0, stream>>>(hbf, Whh1b, Xg, G, t, G4H, H_);
    lstm_gate_kernel<<<gateBlocks, 256, 0, stream>>>(G, c, hbf, nullptr, H1f, t);
  }

  // ---- Projection + log_softmax ----
  logits_logsoftmax_kernel<<<(BS + 255) / 256, 256, 0, stream>>>(H1f, Wout, bout, logp);

  // ---- CRF loss and decode ----
  float* out = (float*)d_out;
  crf_nll_kernel<<<1, B_ * T_, 0, stream>>>(logp, labels, amask,
                                            startt, endt, trans, out);
  crf_viterbi_kernel<<<1, B_ * T_, 0, stream>>>(logp, amask, startt, endt, trans,
                                                hist, out + 1);
}
